// QKVMultiheadAttention_146028888613
// MI455X (gfx1250) — compile-verified, matrix operand order checked by reading, not run on hardware
//
#include <hip/hip_runtime.h>

typedef __attribute__((ext_vector_type(16))) _Float16 v16h;
typedef __attribute__((ext_vector_type(8)))  _Float16 v8h;
typedef __attribute__((ext_vector_type(4)))  _Float16 v4h;
typedef __attribute__((ext_vector_type(8)))  float    v8f;
typedef __attribute__((ext_vector_type(4)))  float    v4f;

#define HEADS   16
#define CH      64
#define NCTX    2048
#define WTOK    (3 * HEADS * CH)   // 3072 floats per token
#define QTILE   128                // query rows per block
#define KTILE   128                // keys per streaming tile
#define NCHUNK  (KTILE / 16)       // 8 S-tiles of 16 keys
#define KFRAG   (KTILE / 32)       // 4 A/B fragment pairs along PV K-dim
#define NWAVES  8

// ---- 16-lane xor-butterfly reductions: pure VALU via v_permlane16_b32 ------
#if defined(__has_builtin)
#  if __has_builtin(__builtin_amdgcn_permlane16)
#    define HAVE_PERMLANE16 1
#  endif
#endif

__device__ __forceinline__ float lanexor16(float v, unsigned s0, unsigned s1)
{
#ifdef HAVE_PERMLANE16
    unsigned u = __float_as_uint(v);
    unsigned p = __builtin_amdgcn_permlane16(u, u, s0, s1, false, false);
    return __uint_as_float(p);
#else
    (void)s0; (void)s1;
    return v;   // unused in fallback path
#endif
}

__device__ __forceinline__ float redmax16(float v)
{
#ifdef HAVE_PERMLANE16
    v = fmaxf(v, lanexor16(v, 0x67452301u, 0xEFCDAB89u)); // ^1
    v = fmaxf(v, lanexor16(v, 0x54761032u, 0xDCFE98BAu)); // ^2
    v = fmaxf(v, lanexor16(v, 0x32107654u, 0xBA98FEDCu)); // ^4
    v = fmaxf(v, lanexor16(v, 0xFEDCBA98u, 0x76543210u)); // ^8
#else
    for (int m = 1; m <= 8; m <<= 1) v = fmaxf(v, __shfl_xor(v, m, 32));
#endif
    return v;
}

__device__ __forceinline__ float redsum16(float v)
{
#ifdef HAVE_PERMLANE16
    v += lanexor16(v, 0x67452301u, 0xEFCDAB89u);
    v += lanexor16(v, 0x54761032u, 0xDCFE98BAu);
    v += lanexor16(v, 0x32107654u, 0xBA98FEDCu);
    v += lanexor16(v, 0xFEDCBA98u, 0x76543210u);
#else
    for (int m = 1; m <= 8; m <<= 1) v += __shfl_xor(v, m, 32);
#endif
    return v;
}

// ---------------------------------------------------------------------------
// Flash-attention forward, one (batch, head, 128-row q-tile) per block.
// 8 waves x 16 q-rows each. All matmuls on v_wmma_f32_16x16x32_f16.
// ---------------------------------------------------------------------------
__global__ __launch_bounds__(256)
void fa_fwd_kernel(const float* __restrict__ qkv, float* __restrict__ out)
{
    __shared__ __align__(16) _Float16 sK [KTILE][CH];         // [key][ch] row-major
    __shared__ __align__(16) _Float16 sVt[CH][KTILE];         // [ch][key] transposed
    __shared__ __align__(16) _Float16 sP [NWAVES][16][KTILE]; // per-wave P scratch

    const int tid   = threadIdx.x;
    const int wave  = tid >> 5;
    const int lane  = tid & 31;
    const int l16   = lane & 15;
    const int half  = lane >> 4;        // 0: lanes 0-15, 1: lanes 16-31
    const int kbase = half * 8;         // 16-bit A/B fragment K-pattern

    const int qtile = blockIdx.x;       // 0..15
    const int head  = blockIdx.y;       // 0..15
    const int batch = blockIdx.z;

    const float scale2 = 0.125f;        // (1/d^(1/4))^2 = 1/sqrt(64)

    // ------- load this wave's Q fragments (16 rows x 64 ch as 2 A-frags) ----
    const int    qrow = qtile * QTILE + wave * 16 + l16;
    const float* qptr = qkv + ((size_t)(batch * NCTX + qrow) * HEADS + head) * (3 * CH);

    v16h qfrag[2];
#pragma unroll
    for (int f = 0; f < 2; ++f) {
        const int k0 = f * 32 + kbase;
        v4f a = *(const v4f*)(qptr + k0);
        v4f b = *(const v4f*)(qptr + k0 + 4);
        v4f c = *(const v4f*)(qptr + k0 + 16);
        v4f d = *(const v4f*)(qptr + k0 + 20);
#pragma unroll
        for (int i = 0; i < 4; ++i) {
            qfrag[f][i]      = (_Float16)(a[i] * scale2);
            qfrag[f][4 + i]  = (_Float16)(b[i] * scale2);
            qfrag[f][8 + i]  = (_Float16)(c[i] * scale2);
            qfrag[f][12 + i] = (_Float16)(d[i] * scale2);
        }
    }

    // ------- online softmax state (lane element r <-> row r + 8*half) ------
    float m[8], l[8];
    v8f   o[4];
#pragma unroll
    for (int r = 0; r < 8; ++r) { m[r] = -1e30f; l[r] = 0.0f; }
#pragma unroll
    for (int cc = 0; cc < 4; ++cc)
#pragma unroll
        for (int r = 0; r < 8; ++r) o[cc][r] = 0.0f;

    // ------- stream over key tiles ------------------------------------------
    for (int kt = 0; kt < NCTX / KTILE; ++kt) {
        __syncthreads();   // previous tile fully consumed

        // cooperative fp32 -> fp16 load: K row-major, V transposed
        {
            const int ch4  = (tid & 15) * 4;  // 0..60
            const int key0 = tid >> 4;        // 0..15
#pragma unroll
            for (int it = 0; it < KTILE / 16; ++it) {
                const int    key = key0 + it * 16;
                const float* kp  = qkv + ((size_t)(batch * NCTX + kt * KTILE + key) * HEADS + head) * (3 * CH) + CH;
                v4f kv = *(const v4f*)(kp + ch4);        // K at +64
                v4f vv = *(const v4f*)(kp + CH + ch4);   // V at +128
                v4h kh;
#pragma unroll
                for (int i = 0; i < 4; ++i) kh[i] = (_Float16)kv[i];
                *(v4h*)&sK[key][ch4] = kh;
#pragma unroll
                for (int i = 0; i < 4; ++i) sVt[ch4 + i][key] = (_Float16)vv[i];
            }
        }

        // prefetch next key tile into cache while we compute
        if (kt + 1 < NCTX / KTILE) {
            const float* np = qkv + ((size_t)(batch * NCTX + (kt + 1) * KTILE + (tid >> 4)) * HEADS + head) * (3 * CH) + CH;
            __builtin_prefetch(np, 0, 0);
        }

        __syncthreads();   // tile resident in LDS

        // ---- S = (Q*s) (K*s)^T : NCHUNK tiles of 16 keys, 2 WMMAs each ----
        v8f s[NCHUNK];
#pragma unroll
        for (int c = 0; c < NCHUNK; ++c) {
            v8f acc;
#pragma unroll
            for (int r = 0; r < 8; ++r) acc[r] = 0.0f;
            const _Float16* kb = &sK[c * 16 + l16][0];
#pragma unroll
            for (int f = 0; f < 2; ++f) {
                v8h lo = *(const v8h*)(kb + f * 32 + kbase);
                v8h hi = *(const v8h*)(kb + f * 32 + kbase + 16);
                v16h bfrag;
#pragma unroll
                for (int i = 0; i < 8; ++i) { bfrag[i] = lo[i]; bfrag[8 + i] = hi[i]; }
                acc = __builtin_amdgcn_wmma_f32_16x16x32_f16(
                          false, qfrag[f], false, bfrag, (short)0, acc, false, false);
            }
            s[c] = acc;
        }

        // ---- online softmax update ----------------------------------------
        float rmax[8];
#pragma unroll
        for (int r = 0; r < 8; ++r) {
            float v = s[0][r];
#pragma unroll
            for (int c = 1; c < NCHUNK; ++c) v = fmaxf(v, s[c][r]);
            rmax[r] = redmax16(v);
        }

        float alpha[8];
#pragma unroll
        for (int r = 0; r < 8; ++r) {
            const float nm = fmaxf(m[r], rmax[r]);
            alpha[r] = __expf(m[r] - nm);
            m[r]     = nm;
        }
#pragma unroll
        for (int cc = 0; cc < 4; ++cc)
#pragma unroll
            for (int r = 0; r < 8; ++r) o[cc][r] *= alpha[r];

        float rsum[8];
#pragma unroll
        for (int r = 0; r < 8; ++r) rsum[r] = 0.0f;
#pragma unroll
        for (int c = 0; c < NCHUNK; ++c)
#pragma unroll
            for (int r = 0; r < 8; ++r) {
                const float p = __expf(s[c][r] - m[r]);
                s[c][r]  = p;
                rsum[r] += p;
            }
#pragma unroll
        for (int r = 0; r < 8; ++r)
            l[r] = l[r] * alpha[r] + redsum16(rsum[r]);

        // ---- spill P (fp16) to per-wave LDS scratch (C-layout -> A-layout) -
#pragma unroll
        for (int c = 0; c < NCHUNK; ++c)
#pragma unroll
            for (int r = 0; r < 8; ++r)
                sP[wave][r + half * 8][c * 16 + l16] = (_Float16)s[c][r];

        asm volatile("s_wait_dscnt 0" ::: "memory");   // wave-local LDS RAW fence

        // ---- O += P V : A from sP, B from transposed V --------------------
        v16h pfrag[KFRAG];
        const _Float16* pb = &sP[wave][l16][0];
#pragma unroll
        for (int f = 0; f < KFRAG; ++f) {
            v8h lo = *(const v8h*)(pb + f * 32 + kbase);
            v8h hi = *(const v8h*)(pb + f * 32 + kbase + 16);
#pragma unroll
            for (int i = 0; i < 8; ++i) { pfrag[f][i] = lo[i]; pfrag[f][8 + i] = hi[i]; }
        }
#pragma unroll
        for (int cc = 0; cc < 4; ++cc) {
            const _Float16* vb = &sVt[cc * 16 + l16][0];
#pragma unroll
            for (int f = 0; f < KFRAG; ++f) {
                v8h lo = *(const v8h*)(vb + f * 32 + kbase);
                v8h hi = *(const v8h*)(vb + f * 32 + kbase + 16);
                v16h bfrag;
#pragma unroll
                for (int i = 0; i < 8; ++i) { bfrag[i] = lo[i]; bfrag[8 + i] = hi[i]; }
                o[cc] = __builtin_amdgcn_wmma_f32_16x16x32_f16(
                            false, pfrag[f], false, bfrag, (short)0, o[cc], false, false);
            }
        }
    }

    // ------- epilogue: O / l -> out[b, t, h*64 + ch] ------------------------
    float inv[8];
#pragma unroll
    for (int r = 0; r < 8; ++r) inv[r] = 1.0f / l[r];
#pragma unroll
    for (int cc = 0; cc < 4; ++cc)
#pragma unroll
        for (int r = 0; r < 8; ++r) {
            const int row = qtile * QTILE + wave * 16 + r + half * 8;
            const int col = cc * 16 + l16;
            out[((size_t)(batch * NCTX + row) * HEADS + head) * CH + col] = o[cc][r] * inv[r];
        }
}

extern "C" void kernel_launch(void* const* d_in, const int* in_sizes, int n_in,
                              void* d_out, int out_size, void* d_ws, size_t ws_size,
                              hipStream_t stream)
{
    const float* qkv = (const float*)d_in[0];
    float*       out = (float*)d_out;
    const int bs = in_sizes[0] / (NCTX * WTOK);   // 2 for the reference shapes

    dim3 grid(NCTX / QTILE, HEADS, bs);
    fa_fwd_kernel<<<grid, 256, 0, stream>>>(qkv, out);
}
